// GAT_89996744720496
// MI455X (gfx1250) — matrix operand order repaired and checked by reading, not compiled, not run on hardware
//
#include <hip/hip_runtime.h>

typedef __attribute__((ext_vector_type(16))) __bf16         v16bf;
typedef __attribute__((ext_vector_type(16))) unsigned short v16us;
typedef __attribute__((ext_vector_type(8)))  unsigned short v8us;
typedef __attribute__((ext_vector_type(8)))  float          v8f;
typedef __attribute__((ext_vector_type(4)))  float          v4f;

#define DEV static __device__ __forceinline__

DEV unsigned short f2bf(float f) {            // fp32 -> bf16, round-to-nearest-even
  unsigned u = __builtin_bit_cast(unsigned, f);
  u += 0x7FFFu + ((u >> 16) & 1u);
  return (unsigned short)(u >> 16);
}
DEV float bf2f(unsigned short u) {
  unsigned x = ((unsigned)u) << 16;
  return __builtin_bit_cast(float, x);
}
DEV v8f vzero8() {
  v8f z;
#pragma unroll
  for (int i = 0; i < 8; ++i) z[i] = 0.f;
  return z;
}

// CDNA5 async global->LDS copy (ASYNCcnt-tracked, bypasses VGPR data path).
DEV void async_b128(const void* lds_dst, const void* gsrc) {
  unsigned lds = (unsigned)(unsigned long long)lds_dst;   // LDS byte offset
  unsigned long long g = (unsigned long long)gsrc;
  asm volatile("global_load_async_to_lds_b128 %0, %1, off"
               :: "v"(lds), "v"(g) : "memory");
}
DEV void wait_async0() { asm volatile("s_wait_asynccnt 0" ::: "memory"); }

DEV v8f wmma_bf16(v16us a16, v16us b16, v8f c) {
  v16bf a = __builtin_bit_cast(v16bf, a16);
  v16bf b = __builtin_bit_cast(v16bf, b16);
  return __builtin_amdgcn_wmma_f32_16x16x32_bf16(false, a, false, b, (short)0, c,
                                                 false, false);
}

// A fragment (16x32 bf16) from row-major [row][k] buffer (LDS or global).
// Lanes 0-15: K = h*8..+7 and 16+h*8..+7 contiguous runs (h = lane>>4).
DEV v16us afrag(const unsigned short* base, int rowStride, int kc) {
  const int lane = (int)(threadIdx.x & 31u);
  const int m = lane & 15, hh = lane >> 4;
  const unsigned short* p = base + m * rowStride + kc + hh * 8;
  v8us lo = *(const v8us*)(p);
  v8us hi = *(const v8us*)(p + 16);
  v16us r;
#pragma unroll
  for (int i = 0; i < 8; ++i) { r[i] = lo[i]; r[i + 8] = hi[i]; }
  return r;
}

// B fragment (32x16 bf16) from K-transposed tile Bt[n][k] (k contiguous).
// Lanes 0-15: K = h*8..+7 and 16+h*8..+7 contiguous runs (h = lane>>4), column n.
DEV v16us bfragT(const unsigned short* Bt, int rowStride, int col0, int kc) {
  const int lane = (int)(threadIdx.x & 31u);
  const int n = lane & 15, hh = lane >> 4;
  const unsigned short* p = Bt + (col0 + n) * rowStride + kc + hh * 8;
  v8us lo = *(const v8us*)(p);
  v8us hi = *(const v8us*)(p + 16);
  v16us r;
#pragma unroll
  for (int i = 0; i < 8; ++i) { r[i] = lo[i]; r[i + 8] = hi[i]; }
  return r;
}

// ===========================================================================
// Prep kernels: one-shot fp32->bf16 conversion / transposition of weights
// ===========================================================================
__global__ __launch_bounds__(256) void p_cvt(const float* __restrict__ in,
                                             unsigned short* __restrict__ out,
                                             int n, float scale) {
  const int i = blockIdx.x * 256 + threadIdx.x;
  if (i < n) out[i] = f2bf(in[i] * scale);
}

// Generic 64x64-tiled transpose: out[n*ostride + k] = bf16(in[k*N + n])
__global__ __launch_bounds__(256) void p_tr64(const float* __restrict__ in,
                                              unsigned short* __restrict__ out,
                                              int N, long long ostride) {
  __shared__ __align__(16) unsigned short tile[64 * 72];
  const int kb = blockIdx.x * 64, nb = blockIdx.y * 64;
  const int r = threadIdx.x >> 2, q = threadIdx.x & 3;
  {
    const float* src = in + (long long)(kb + r) * N + nb + q * 16;
#pragma unroll
    for (int j = 0; j < 4; ++j) {
      v4f f = *(const v4f*)(src + j * 4);
#pragma unroll
      for (int u = 0; u < 4; ++u) tile[r * 72 + q * 16 + j * 4 + u] = f2bf(f[u]);
    }
  }
  __syncthreads();
  {
    v8us o0, o1;
#pragma unroll
    for (int j = 0; j < 8; ++j) {
      o0[j] = tile[(q * 16 + j) * 72 + r];
      o1[j] = tile[(q * 16 + 8 + j) * 72 + r];
    }
    unsigned short* dst = out + (long long)(nb + r) * ostride + kb + q * 16;
    *(v8us*)dst = o0;
    *(v8us*)(dst + 8) = o1;
  }
}

// W_se segment transpose: WseT[n][sm*64 + k] = bf16(Wse[s*195+mod*65+k][n])
__global__ __launch_bounds__(256) void p_wseg(const float* __restrict__ Wse,
                                              unsigned short* __restrict__ WseT) {
  __shared__ __align__(16) unsigned short tile[64 * 72];
  const int sm = blockIdx.x;
  const int s = sm / 3, mod = sm - 3 * s;
  const int wbase = s * 195 + mod * 65;
  const int nb = blockIdx.y * 64;
  const int r = threadIdx.x >> 2, q = threadIdx.x & 3;
  {
    const float* src = Wse + (long long)(wbase + r) * 384 + nb + q * 16;
#pragma unroll
    for (int j = 0; j < 4; ++j) {
      v4f f = *(const v4f*)(src + j * 4);
#pragma unroll
      for (int u = 0; u < 4; ++u) tile[r * 72 + q * 16 + j * 4 + u] = f2bf(f[u]);
    }
  }
  __syncthreads();
  {
    v8us o0, o1;
#pragma unroll
    for (int j = 0; j < 8; ++j) {
      o0[j] = tile[(q * 16 + j) * 72 + r];
      o1[j] = tile[(q * 16 + 8 + j) * 72 + r];
    }
    unsigned short* dst = WseT + (long long)(nb + r) * 19200 + sm * 64 + q * 16;
    *(v8us*)dst = o0;
    *(v8us*)(dst + 8) = o1;
  }
}

// Count-row gather: WcT[n*320 + k] = bf16(Wse[s*195+mod*65+64][n]), k<300
__global__ __launch_bounds__(256) void p_wct(const float* __restrict__ Wse,
                                             unsigned short* __restrict__ WcT) {
  const int o = blockIdx.x * 256 + threadIdx.x;   // 384*320
  const int n = o / 320, k = o - n * 320;
  float v = 0.f;
  if (k < 300) {
    const int s = k / 3, mod = k - 3 * s;
    v = Wse[(long long)(s * 195 + mod * 65 + 64) * 384 + n];
  }
  WcT[o] = f2bf(v);
}

// ===========================================================================
// K1: fusion_r = [gather(emb)|counts] @ W_se + b_se   (8192 x 19500 x 384)
// 64-row block tile; each wave owns 16 rows x 192 cols (12 accumulators);
// 24 WMMAs per barrier pair; B-fragments software-pipelined one tile ahead.
// ===========================================================================
__global__ __launch_bounds__(256) void k1_fusion(
    const int* __restrict__ le, const int* __restrict__ me, const int* __restrict__ te,
    const unsigned short* __restrict__ cb0, const unsigned short* __restrict__ cb1,
    const unsigned short* __restrict__ cb2,
    const unsigned short* __restrict__ embb, const unsigned short* __restrict__ WseT,
    const unsigned short* __restrict__ WcT, const float* __restrict__ bse,
    float* __restrict__ fusf, unsigned short* __restrict__ fusb)
{
  __shared__ __align__(16) unsigned short As[64 * 72];   // 64 rows x 64 k
  __shared__ __align__(16) unsigned short Bt[384 * 72];  // 384 cols x 64 k
  const int tid = threadIdx.x, wave = tid >> 5, lane = tid & 31;
  const int row0 = blockIdx.x * 64;
  const int rowsub = wave >> 1;                 // 0..3 (16 rows each)
  const int colbase = (wave & 1) * 192;         // 2 wave-groups cover 384 cols
  v8f acc[12];
#pragma unroll
  for (int t = 0; t < 12; ++t) acc[t] = vzero8();

  // ---- embedding K: 300 (s,mod) segments of K=64 ----
  for (int sm = 0; sm < 300; ++sm) {
    const int s = sm / 3, mod = sm - 3 * s;
    {   // async staging: pure 16B copies, no VALU data path
      const int* ip = (mod == 0) ? le : ((mod == 1) ? me : te);
#pragma unroll
      for (int i = 0; i < 2; ++i) {             // A: 64 rows x 8 chunks
        const int cid = tid + 256 * i;
        const int r = cid >> 3, q8 = (cid & 7) * 8;
        const int idx = ip[(row0 + r) * 100 + s];
        async_b128(&As[r * 72 + q8], embb + (long long)idx * 64 + q8);
      }
#pragma unroll
      for (int i = 0; i < 12; ++i) {            // B: 384 rows x 8 chunks
        const int cid = tid + 256 * i;
        const int n = cid >> 3, c8 = (cid & 7) * 8;
        async_b128(&Bt[n * 72 + c8],
                   WseT + (long long)n * 19200 + sm * 64 + c8);
      }
      if (sm < 299) {  // near-scope prefetch of full next B segment (1 line/row)
        __builtin_prefetch(WseT + (long long)tid * 19200 + (sm + 1) * 64, 0, 3);
        if (tid < 128)
          __builtin_prefetch(WseT + (long long)(256 + tid) * 19200 + (sm + 1) * 64,
                             0, 3);
      }
      wait_async0();
    }
    __syncthreads();
#pragma unroll
    for (int kc = 0; kc < 64; kc += 32) {
      v16us a = afrag(As + rowsub * 16 * 72, 72, kc);
      v16us bcur = bfragT(Bt, 72, colbase, kc);   // one-tile-ahead pipeline
#pragma unroll
      for (int t = 0; t < 12; ++t) {
        v16us bnxt = bcur;
        if (t < 11) bnxt = bfragT(Bt, 72, colbase + (t + 1) * 16, kc);
        acc[t] = wmma_bf16(a, bcur, acc[t]);
        bcur = bnxt;
      }
    }
    __syncthreads();
  }

  // ---- count columns: K = 300 padded to 320 ----
  for (int ck = 0; ck < 10; ++ck) {
    {
      const int ar = tid >> 2, aq = tid & 3;    // 64 rows x 32 k, 8 elems/thread
#pragma unroll
      for (int j = 0; j < 8; ++j) {
        const int k = ck * 32 + aq * 8 + j;
        unsigned short v = 0;
        if (k < 300) {
          const int s = k / 3, mod = k - 3 * s;
          const unsigned short* cp = (mod == 0) ? cb0 : ((mod == 1) ? cb1 : cb2);
          v = cp[(row0 + ar) * 100 + s];
        }
        As[ar * 72 + aq * 8 + j] = v;
      }
#pragma unroll
      for (int i = 0; i < 6; ++i) {             // B: 384 rows x 4 chunks
        const int cid = tid + 256 * i;
        const int n = cid >> 2, c8 = (cid & 3) * 8;
        async_b128(&Bt[n * 72 + c8], WcT + n * 320 + ck * 32 + c8);
      }
      wait_async0();
    }
    __syncthreads();
    {
      v16us a = afrag(As + rowsub * 16 * 72, 72, 0);
      v16us bcur = bfragT(Bt, 72, colbase, 0);
#pragma unroll
      for (int t = 0; t < 12; ++t) {
        v16us bnxt = bcur;
        if (t < 11) bnxt = bfragT(Bt, 72, colbase + (t + 1) * 16, 0);
        acc[t] = wmma_bf16(a, bcur, acc[t]);
        bcur = bnxt;
      }
    }
    __syncthreads();
  }

  {   // epilogue: +b_se, fp32 to d_out, bf16 copy for downstream GEMMs
    const int n = lane & 15, hh = lane >> 4;
#pragma unroll
    for (int t = 0; t < 12; ++t) {
      const int col = colbase + t * 16 + n;
      const float bs = bse[col];
#pragma unroll
      for (int r = 0; r < 8; ++r) {
        const int grow = row0 + rowsub * 16 + hh * 8 + r;
        const float v = acc[t][r] + bs;
        fusf[grow * 384 + col] = v;
        fusb[grow * 384 + col] = f2bf(v);
      }
    }
  }
}

// ===========================================================================
// K2: h = fusion_r @ gat_W   (8192 x 384 x 384); writes h row-major + batchT
// ===========================================================================
__global__ __launch_bounds__(256) void k2_h(
    const unsigned short* __restrict__ fusb, const unsigned short* __restrict__ gWt,
    unsigned short* __restrict__ hb, unsigned short* __restrict__ hbt)
{
  __shared__ __align__(16) unsigned short Bt[384 * 72];
  const int tid = threadIdx.x, wave = tid >> 5, lane = tid & 31;
  const int row0 = blockIdx.x * 32;
  const int rowsub = wave >> 2, colbase = (wave & 3) * 96;
  v8f acc[6];
#pragma unroll
  for (int t = 0; t < 6; ++t) acc[t] = vzero8();

  for (int kb = 0; kb < 384; kb += 64) {
#pragma unroll
    for (int i = 0; i < 12; ++i) {
      const int cid = tid + 256 * i;
      const int n = cid >> 3, c8 = (cid & 7) * 8;
      async_b128(&Bt[n * 72 + c8], gWt + n * 384 + kb + c8);
    }
    wait_async0();
    __syncthreads();
#pragma unroll
    for (int kc = 0; kc < 64; kc += 32) {
      v16us a = afrag(fusb + (row0 + rowsub * 16) * 384, 384, kb + kc);
      v16us bcur = bfragT(Bt, 72, colbase, kc);
#pragma unroll
      for (int t = 0; t < 6; ++t) {
        v16us bnxt = bcur;
        if (t < 5) bnxt = bfragT(Bt, 72, colbase + (t + 1) * 16, kc);
        acc[t] = wmma_bf16(a, bcur, acc[t]);
        bcur = bnxt;
      }
    }
    __syncthreads();
  }
  const int n = lane & 15, hh = lane >> 4;
#pragma unroll
  for (int t = 0; t < 6; ++t) {
    const int col = colbase + t * 16 + n;
#pragma unroll
    for (int r = 0; r < 8; ++r) {
      const int grow = row0 + rowsub * 16 + hh * 8 + r;
      const unsigned short hv = f2bf(acc[t][r]);
      hb[grow * 384 + col] = hv;
      hbt[((grow >> 7) * 384 + col) * 128 + (grow & 127)] = hv;  // [b][feat][node]
    }
  }
}

// ===========================================================================
// K2b: ha1 = h@a1, ha2 = h@a2   (one wave per row)
// ===========================================================================
__global__ __launch_bounds__(256) void k2b_ha(
    const unsigned short* __restrict__ hb, const float* __restrict__ gata,
    float* __restrict__ ha1, float* __restrict__ ha2)
{
  const int g = blockIdx.x * 8 + (threadIdx.x >> 5);
  const int lane = threadIdx.x & 31;
  float s1 = 0.f, s2 = 0.f;
#pragma unroll
  for (int j = 0; j < 12; ++j) {
    const int col = j * 32 + lane;
    const float hv = bf2f(hb[g * 384 + col]);
    s1 += hv * gata[col];
    s2 += hv * gata[384 + col];
  }
#pragma unroll
  for (int off = 16; off > 0; off >>= 1) {
    s1 += __shfl_xor(s1, off);
    s2 += __shfl_xor(s2, off);
  }
  if (lane == 0) { ha1[g] = s1; ha2[g] = s2; }
}

// ===========================================================================
// K3a: attb = bf16(softmax(mask(leakyrelu(ha1_i + ha2_j))))
// ===========================================================================
__global__ __launch_bounds__(256) void k3a_att(
    const float* __restrict__ ha1, const float* __restrict__ ha2,
    const float* __restrict__ adj, unsigned short* __restrict__ attb)
{
  const int g = blockIdx.x * 8 + (threadIdx.x >> 5);
  const int lane = threadIdx.x & 31;
  const int b = g >> 7, i = g & 127;
  const float h1 = ha1[g];
  float e[4];
  float mx = -3.0e38f;
#pragma unroll
  for (int q = 0; q < 4; ++q) {
    const int j = q * 32 + lane;
    float v = h1 + ha2[b * 128 + j];
    v = (v > 0.f) ? v : 0.2f * v;                 // LEAKY_ALPHA
    v = (adj[i * 128 + j] > 0.f) ? v : -1.0e12f;  // mask
    e[q] = v;
    mx = fmaxf(mx, v);
  }
#pragma unroll
  for (int off = 16; off > 0; off >>= 1) mx = fmaxf(mx, __shfl_xor(mx, off));
  float sum = 0.f;
#pragma unroll
  for (int q = 0; q < 4; ++q) { e[q] = __expf(e[q] - mx); sum += e[q]; }
#pragma unroll
  for (int off = 16; off > 0; off >>= 1) sum += __shfl_xor(sum, off);
  const float inv = 1.f / sum;
#pragma unroll
  for (int q = 0; q < 4; ++q) attb[g * 128 + q * 32 + lane] = f2bf(e[q] * inv);
}

// ===========================================================================
// K3b: x1 = fusion_r + relu(att @ h)   (batched 128x128x384)
// ===========================================================================
__global__ __launch_bounds__(256) void k3b_out(
    const unsigned short* __restrict__ attb, const unsigned short* __restrict__ hbt,
    const float* __restrict__ fusf, unsigned short* __restrict__ x1b)
{
  __shared__ __align__(16) unsigned short Bt[384 * 72];
  const int tid = threadIdx.x, wave = tid >> 5, lane = tid & 31;
  const int b = blockIdx.x >> 2, rowin0 = (blockIdx.x & 3) * 32;
  const int rowsub = wave >> 2, colbase = (wave & 3) * 96;
  v8f acc[6];
#pragma unroll
  for (int t = 0; t < 6; ++t) acc[t] = vzero8();

  for (int kb = 0; kb < 128; kb += 64) {
#pragma unroll
    for (int i = 0; i < 12; ++i) {
      const int cid = tid + 256 * i;
      const int n = cid >> 3, c8 = (cid & 7) * 8;
      async_b128(&Bt[n * 72 + c8], hbt + (b * 384 + n) * 128 + kb + c8);
    }
    wait_async0();
    __syncthreads();
#pragma unroll
    for (int kc = 0; kc < 64; kc += 32) {
      v16us a = afrag(attb + (b * 128 + rowin0 + rowsub * 16) * 128, 128, kb + kc);
      v16us bcur = bfragT(Bt, 72, colbase, kc);
#pragma unroll
      for (int t = 0; t < 6; ++t) {
        v16us bnxt = bcur;
        if (t < 5) bnxt = bfragT(Bt, 72, colbase + (t + 1) * 16, kc);
        acc[t] = wmma_bf16(a, bcur, acc[t]);
        bcur = bnxt;
      }
    }
    __syncthreads();
  }
  const int n = lane & 15, hh = lane >> 4;
#pragma unroll
  for (int t = 0; t < 6; ++t) {
    const int col = colbase + t * 16 + n;
#pragma unroll
    for (int r = 0; r < 8; ++r) {
      const int grow = b * 128 + rowin0 + rowsub * 16 + hh * 8 + r;
      const float o = fmaxf(acc[t][r], 0.f);                     // relu
      x1b[grow * 384 + col] = f2bf(fusf[grow * 384 + col] + o);  // residual
    }
  }
}

// ===========================================================================
// K4: 64-way split-K partials of x1(64x49152) @ W1(49152x128)
// ===========================================================================
__global__ __launch_bounds__(256) void k4_partial(
    const unsigned short* __restrict__ x1b, const unsigned short* __restrict__ W1t,
    float* __restrict__ part)
{
  __shared__ __align__(16) unsigned short Bt[128 * 72];
  const int tid = threadIdx.x, wave = tid >> 5, lane = tid & 31;
  const int kstart = blockIdx.x * 768;
  const int rowsub = wave >> 1;            // 4 row-subtiles -> 64 rows
  const int cb = (wave & 1) * 64;          // 2 wave-groups cover 128 cols
  v8f acc[4];
#pragma unroll
  for (int t = 0; t < 4; ++t) acc[t] = vzero8();

  for (int kci = 0; kci < 12; ++kci) {
    const int kb = kstart + kci * 64;
#pragma unroll
    for (int i = 0; i < 4; ++i) {
      const int cid = tid + 256 * i;
      const int n = cid >> 3, c8 = (cid & 7) * 8;
      async_b128(&Bt[n * 72 + c8], W1t + (long long)n * 49152 + kb + c8);
    }
    wait_async0();
    __syncthreads();
#pragma unroll
    for (int kc = 0; kc < 64; kc += 32) {
      v16us a = afrag(x1b + rowsub * 16 * 49152, 49152, kb + kc);
      v16us bcur = bfragT(Bt, 72, cb, kc);
#pragma unroll
      for (int t = 0; t < 4; ++t) {
        v16us bnxt = bcur;
        if (t < 3) bnxt = bfragT(Bt, 72, cb + (t + 1) * 16, kc);
        acc[t] = wmma_bf16(a, bcur, acc[t]);
        bcur = bnxt;
      }
    }
    __syncthreads();
  }
  const int n = lane & 15, hh = lane >> 4;
#pragma unroll
  for (int t = 0; t < 4; ++t) {
    const int col = cb + t * 16 + n;
#pragma unroll
    for (int r = 0; r < 8; ++r) {
      const int grow = rowsub * 16 + hh * 8 + r;
      part[blockIdx.x * 8192 + grow * 128 + col] = acc[t][r];
    }
  }
}

// K4b: deterministic reduction of the 64 partials + b1
__global__ __launch_bounds__(256) void k4b_reduce(
    const float* __restrict__ part, const float* __restrict__ b1,
    float* __restrict__ xo)
{
  const int o = blockIdx.x * 256 + threadIdx.x;   // 8192 outputs
  float s = b1[o & 127];
  for (int sl = 0; sl < 64; ++sl) s += part[sl * 8192 + o];
  xo[o] = s;
}

// ===========================================================================
extern "C" void kernel_launch(void* const* d_in, const int* in_sizes, int n_in,
                              void* d_out, int out_size, void* d_ws, size_t ws_size,
                              hipStream_t stream)
{
  (void)in_sizes; (void)n_in; (void)out_size; (void)ws_size;
  const int*   le   = (const int*)d_in[0];
  const int*   me   = (const int*)d_in[1];
  const int*   te   = (const int*)d_in[2];
  const float* lc   = (const float*)d_in[3];
  const float* mc   = (const float*)d_in[4];
  const float* tc   = (const float*)d_in[5];
  const float* adj  = (const float*)d_in[6];
  const float* emb  = (const float*)d_in[7];
  const float* Wse  = (const float*)d_in[8];
  const float* bse  = (const float*)d_in[9];
  const float* gatW = (const float*)d_in[10];
  const float* gata = (const float*)d_in[11];
  const float* W1   = (const float*)d_in[12];
  const float* b1   = (const float*)d_in[13];

  float* xo   = (float*)d_out;        // x: 64*128
  float* fusf = xo + 8192;            // fusion_r: 8192*384 fp32

  char* ws = (char*)d_ws;
  size_t off = 0;
  auto carve = [&](size_t bytes) {
    void* p = ws + off;
    off += (bytes + 255) & ~(size_t)255;
    return p;
  };
  unsigned short* embb = (unsigned short*)carve(2000ull * 64 * 2);
  unsigned short* cb0  = (unsigned short*)carve(819200ull * 2);
  unsigned short* cb1  = (unsigned short*)carve(819200ull * 2);
  unsigned short* cb2  = (unsigned short*)carve(819200ull * 2);
  unsigned short* WseT = (unsigned short*)carve(384ull * 19200 * 2);
  unsigned short* WcT  = (unsigned short*)carve(384ull * 320 * 2);
  unsigned short* gWt  = (unsigned short*)carve(384ull * 384 * 2);
  unsigned short* W1t  = (unsigned short*)carve(128ull * 49152 * 2);
  unsigned short* fusb = (unsigned short*)carve(8192ull * 384 * 2);
  unsigned short* hb   = (unsigned short*)carve(8192ull * 384 * 2);
  unsigned short* hbt  = (unsigned short*)carve(8192ull * 384 * 2);
  unsigned short* attb = (unsigned short*)carve(64ull * 128 * 128 * 2);
  unsigned short* x1b  = (unsigned short*)carve(8192ull * 384 * 2);
  float* part = (float*)carve(64ull * 8192 * 4);
  float* ha1  = (float*)carve(8192ull * 4);
  float* ha2  = (float*)carve(8192ull * 4);

  // one-shot weight conversion / transposition
  p_cvt<<<500, 256, 0, stream>>>(emb, embb, 128000, 1.0f);
  p_cvt<<<3200, 256, 0, stream>>>(lc, cb0, 819200, 0.01f);
  p_cvt<<<3200, 256, 0, stream>>>(mc, cb1, 819200, 0.01f);
  p_cvt<<<3200, 256, 0, stream>>>(tc, cb2, 819200, 0.01f);
  p_wseg<<<dim3(300, 6), 256, 0, stream>>>(Wse, WseT);
  p_wct<<<480, 256, 0, stream>>>(Wse, WcT);
  p_tr64<<<dim3(6, 6), 256, 0, stream>>>(gatW, gWt, 384, 384);
  p_tr64<<<dim3(768, 2), 256, 0, stream>>>(W1, W1t, 128, 49152);

  // main pipeline
  k1_fusion <<<128, 256, 0, stream>>>(le, me, te, cb0, cb1, cb2, embb, WseT, WcT,
                                      bse, fusf, fusb);
  k2_h      <<<256, 256, 0, stream>>>(fusb, gWt, hb, hbt);
  k2b_ha    <<<1024, 256, 0, stream>>>(hb, gata, ha1, ha2);
  k3a_att   <<<1024, 256, 0, stream>>>(ha1, ha2, adj, attb);
  k3b_out   <<<256, 256, 0, stream>>>(attb, hbt, fusf, x1b);
  k4_partial<<<64, 256, 0, stream>>>(x1b, W1t, part);
  k4b_reduce<<<32, 256, 0, stream>>>(part, b1, xo);
}
